// SFC_CAE_Encoder_47313359733332
// MI455X (gfx1250) — matrix operand order repaired and checked by reading, not compiled, not run on hardware
//
#include <hip/hip_runtime.h>
#include <hip/hip_bf16.h>
#include <math.h>

typedef __attribute__((ext_vector_type(16))) _Float16 v16h;
typedef __attribute__((ext_vector_type(8)))  float    v8f;

#define SFC    2
#define BATCH  64
#define NPTS   65536

// ---------------------------------------------------------------------------
// Stage A: gather + 3-tap weighted sum + tanh -> a0[i][b][n]  (stored f16)
// ---------------------------------------------------------------------------
__global__ void stage_a_kernel(const float* __restrict__ x,
                               const int* __restrict__ ordi,
                               const int* __restrict__ mini,
                               const int* __restrict__ plui,
                               const float* __restrict__ nnw,
                               const float* __restrict__ nnb,
                               _Float16* __restrict__ a0) {
  size_t gid = (size_t)blockIdx.x * blockDim.x + threadIdx.x;
  const size_t total = (size_t)SFC * BATCH * NPTS;   // 2^23
  if (gid >= total) return;
  int n = (int)(gid & (NPTS - 1));
  int b = (int)((gid >> 16) & (BATCH - 1));
  int i = (int)(gid >> 22);
  const float* xb = x + (size_t)b * NPTS;
  int io = i * NPTS + n;
  float tm = xb[mini[io]];
  float to = xb[ordi[io]];
  float tp = xb[plui[io]];
  const float* wv = nnw + (size_t)io * 3;
  float s = tm * wv[0] + to * wv[1] + tp * wv[2] + nnb[io];
  a0[gid] = (_Float16)tanhf(s);   // gid == ((i*64+b)*65536 + n)
}

// ---------------------------------------------------------------------------
// Conv-as-GEMM via v_wmma_f32_16x16x32_f16.
//   M = 16 output positions, N = 16 output channels (zero-padded if COUT<16),
//   K = CIN*32 taken one input channel (32 taps) per WMMA step.
// in  : [SFC][BATCH][CIN][INLEN]  f16 activations
// w   : [SFC][COUT][CIN][32] f32, bias: [SFC][COUT] f32
// out : f16; HMODE ? h[b][i*16400 + oc*OUTLEN + t] : [SFC][BATCH][COUT][OUTLEN]
// ---------------------------------------------------------------------------
template<int CIN, int COUT, int INLEN, int OUTLEN, bool HMODE>
__global__ void conv_wmma_kernel(const _Float16* __restrict__ in,
                                 const float* __restrict__ w,
                                 const float* __restrict__ bias,
                                 _Float16* __restrict__ out) {
  __shared__ _Float16 ldsw[16 * CIN * 32];       // [n(16)][cin][32] as f16
  const int i = blockIdx.z;
  const int b = blockIdx.y;
  const float* wi = w + (size_t)i * COUT * CIN * 32;
  for (int idx = threadIdx.x; idx < 16 * CIN * 32; idx += blockDim.x) {
    int oc = idx / (CIN * 32);
    int r  = idx % (CIN * 32);
    ldsw[idx] = (oc < COUT) ? (_Float16)wi[(size_t)oc * CIN * 32 + r]
                            : (_Float16)0.0f;
  }
  __syncthreads();

  const int lane = threadIdx.x & 31;
  const int wave = threadIdx.x >> 5;
  const int t0 = (blockIdx.x * (blockDim.x >> 5) + wave) * 16;
  if (t0 >= OUTLEN) return;                       // wave-uniform

  const int m  = lane & 15;                       // A row / D column index
  const int hh = lane >> 4;                       // half-wave
  const int t  = t0 + m;                          // output position of A row
  const _Float16* inb = in + ((size_t)(i * BATCH + b)) * CIN * INLEN;
  const int pbase = t * 4 - 16;                   // STRIDE=4, PAD=16
  // interior iff every lane's window [pbase, pbase+31] is inside [0, INLEN)
  const bool interior = (t0 >= 4) && (4 * t0 + 75 < INLEN);  // wave-uniform

  v8f acc = {};
  for (int ic = 0; ic < CIN; ++ic) {
    const _Float16* inc = inb + (size_t)ic * INLEN;
    v16h a;
    if (interior) {                              // unguarded -> wide loads
#pragma unroll
      for (int e = 0; e < 16; ++e) {
        int kw = 8 * hh + e + ((e >= 8) ? 8 : 0);
        a[e] = inc[pbase + kw];
      }
    } else {                                     // exact zero-padding path
#pragma unroll
      for (int e = 0; e < 16; ++e) {
        int kw  = 8 * hh + e + ((e >= 8) ? 8 : 0);
        int pos = pbase + kw;
        a[e] = (pos >= 0 && pos < INLEN) ? inc[pos] : (_Float16)0.0f;
      }
    }
    v16h bf;
    const _Float16* lw = ldsw + ((size_t)m * CIN + ic) * 32;
#pragma unroll
    for (int e = 0; e < 16; ++e) {
      bf[e] = lw[8 * hh + e + ((e >= 8) ? 8 : 0)]; // B 32x16 layout: k = 8h+e (+8 for e>=8)
    }
    acc = __builtin_amdgcn_wmma_f32_16x16x32_f16(false, a, false, bf,
                                                 (short)0, acc, false, false);
  }

  const int oc = lane & 15;                      // D column = out channel
  if (oc < COUT) {
    float bv = bias[i * COUT + oc];
#pragma unroll
    for (int v = 0; v < 8; ++v) {
      int tt = t0 + v + 8 * hh;                  // D row v -> M = v + 8h
      if (tt < OUTLEN) {
        _Float16 r = (_Float16)tanhf(acc[v] + bv);
        if (HMODE) {
          out[(size_t)b * 32800 + (size_t)i * 16400 + (size_t)oc * OUTLEN + tt] = r;
        } else {
          out[(((size_t)(i * BATCH + b)) * COUT + oc) * OUTLEN + tt] = r;
        }
      }
    }
  }
}

// ---------------------------------------------------------------------------
// fc1: (64 x 32800) @ (32800 x 1024), K split into 41 chunks of 800.
// Each wave computes the full M=64 column strip (4 accumulators) for one
// (n-tile, k-chunk): fw1 fetched from HBM exactly once; B fragment reused
// by 4 back-to-back WMMAs; A fragments preloaded so loads overlap.
// partials[kc][64][1024] fp32 (deterministic split-K).
// ---------------------------------------------------------------------------
__global__ void fc1_wmma_kernel(const _Float16* __restrict__ hbuf,
                                const float* __restrict__ fw1,
                                float* __restrict__ partials) {
  const int lane = threadIdx.x & 31;
  const int wave = threadIdx.x >> 5;
  const int nt = blockIdx.x * 8 + wave;          // 0..63 (N tiles)
  const int kc = blockIdx.y;                     // 0..40 (K chunks)
  const int m  = lane & 15;
  const int hh = lane >> 4;
  const int ncol = nt * 16 + m;

  v8f acc[4] = {v8f{}, v8f{}, v8f{}, v8f{}};
  int k0 = kc * 800;                             // 25 WMMA steps of K=32
  for (int s = 0; s < 25; ++s, k0 += 32) {
    if (s < 23)                                  // stream-prefetch fw1 rows
      __builtin_prefetch(fw1 + (size_t)(k0 + 64 + 16 * hh) * 1024 + ncol, 0, 1);
    // B fragment: strided fp32 loads, packed to f16
    v16h bf;
#pragma unroll
    for (int e = 0; e < 16; ++e) {
      int kk = k0 + 8 * hh + e + ((e >= 8) ? 8 : 0);
      bf[e] = (_Float16)fw1[(size_t)kk * 1024 + ncol];
    }
    // Preload all 4 A fragments (f16 rows, merge into wide loads)
    v16h av[4];
#pragma unroll
    for (int mt = 0; mt < 4; ++mt) {
      const _Float16* arow = hbuf + (size_t)(mt * 16 + m) * 32800;
#pragma unroll
      for (int e = 0; e < 16; ++e) {
        int kw = 8 * hh + e + ((e >= 8) ? 8 : 0);
        av[mt][e] = arow[k0 + kw];
      }
    }
#pragma unroll
    for (int mt = 0; mt < 4; ++mt) {
      acc[mt] = __builtin_amdgcn_wmma_f32_16x16x32_f16(false, av[mt], false, bf,
                                                       (short)0, acc[mt],
                                                       false, false);
    }
  }
  float* pout = partials + (size_t)kc * 64 * 1024;
#pragma unroll
  for (int mt = 0; mt < 4; ++mt) {
#pragma unroll
    for (int v = 0; v < 8; ++v) {
      pout[(size_t)(mt * 16 + v + 8 * hh) * 1024 + ncol] = acc[mt][v];
    }
  }
}

__global__ void fc1_reduce_kernel(const float* __restrict__ partials,
                                  const float* __restrict__ fb1,
                                  _Float16* __restrict__ h1) {
  int gid = blockIdx.x * blockDim.x + threadIdx.x;   // 65536
  if (gid >= 64 * 1024) return;
  float s = fb1[gid & 1023];
  for (int kc = 0; kc < 41; ++kc) s += partials[(size_t)kc * 65536 + gid];
  h1[gid] = (_Float16)tanhf(s);
}

// ---------------------------------------------------------------------------
// fc2: (64 x 1024) @ (1024 x 16) + tanh -> d_out (64 x 16) fp32. 4 waves.
// ---------------------------------------------------------------------------
__global__ void fc2_wmma_kernel(const _Float16* __restrict__ h1,
                                const float* __restrict__ fw2,
                                const float* __restrict__ fb2,
                                float* __restrict__ out) {
  const int lane = threadIdx.x & 31;
  const int mt = threadIdx.x >> 5;                // 0..3
  const int m  = lane & 15;
  const int hh = lane >> 4;
  const int ncol = m;                             // N = 16
  const _Float16* arow = h1 + (size_t)(mt * 16 + m) * 1024;
  v8f acc = {};
  for (int k0 = 0; k0 < 1024; k0 += 32) {
    v16h a, bf;
#pragma unroll
    for (int e = 0; e < 16; ++e) {
      int kw = 8 * hh + e + ((e >= 8) ? 8 : 0);
      a[e] = arow[k0 + kw];
    }
#pragma unroll
    for (int e = 0; e < 16; ++e) {
      bf[e] = (_Float16)fw2[(size_t)(k0 + 8 * hh + e + ((e >= 8) ? 8 : 0)) * 16 + ncol];
    }
    acc = __builtin_amdgcn_wmma_f32_16x16x32_f16(false, a, false, bf,
                                                 (short)0, acc, false, false);
  }
  float bv = fb2[ncol];
#pragma unroll
  for (int v = 0; v < 8; ++v) {
    out[(size_t)(mt * 16 + v + 8 * hh) * 16 + ncol] = tanhf(acc[v] + bv);
  }
}

// ---------------------------------------------------------------------------
// Workspace layout (bytes), f16 activations:
//   a0    @ 0          : 2*64*65536*2    = 16,777,216
//   o1    @ 16777216   : 2*64*8*16385*2  = 33,556,480
//   o2    @ 50333696   : 2*64*16*4097*2  = 16,781,312
//   hbuf  @ 67115008   : 64*32800*2      =  4,198,400
//   part  @ 71313408   : 41*64*1024*4    = 10,747,904  (fp32)
//   h1    @ 82061312   : 64*1024*2       =    131,072  (end 82,192,384)
// ---------------------------------------------------------------------------
extern "C" void kernel_launch(void* const* d_in, const int* in_sizes, int n_in,
                              void* d_out, int out_size, void* d_ws, size_t ws_size,
                              hipStream_t stream) {
  const float* x    = (const float*)d_in[0];
  const int*   ordi = (const int*)d_in[1];
  const int*   mini = (const int*)d_in[2];
  const int*   plui = (const int*)d_in[3];
  const float* nnw  = (const float*)d_in[4];
  const float* nnb  = (const float*)d_in[5];
  const float* w1   = (const float*)d_in[6];
  const float* b1   = (const float*)d_in[7];
  const float* w2   = (const float*)d_in[8];
  const float* b2   = (const float*)d_in[9];
  const float* w3   = (const float*)d_in[10];
  const float* b3   = (const float*)d_in[11];
  const float* fw1  = (const float*)d_in[12];
  const float* fb1  = (const float*)d_in[13];
  const float* fw2  = (const float*)d_in[14];
  const float* fb2  = (const float*)d_in[15];

  char* ws = (char*)d_ws;
  _Float16* a0   = (_Float16*)(ws + 0);
  _Float16* o1   = (_Float16*)(ws + 16777216);
  _Float16* o2   = (_Float16*)(ws + 50333696);
  _Float16* hbuf = (_Float16*)(ws + 67115008);
  float*    part = (float*)   (ws + 71313408);
  _Float16* h1   = (_Float16*)(ws + 82061312);

  // Stage A: 2*64*65536 = 8,388,608 elements
  stage_a_kernel<<<32768, 256, 0, stream>>>(x, ordi, mini, plui, nnw, nnb, a0);

  // conv1: 16385 outputs -> 1025 tiles of 16 -> ceil(1025/8)=129 blocks
  conv_wmma_kernel<1, 8, 65536, 16385, false>
      <<<dim3(129, BATCH, SFC), 256, 0, stream>>>(a0, w1, b1, o1);
  // conv2: 4097 outputs -> 257 tiles -> 33 blocks
  conv_wmma_kernel<8, 16, 16385, 4097, false>
      <<<dim3(33, BATCH, SFC), 256, 0, stream>>>(o1, w2, b2, o2);
  // conv3: 1025 outputs -> 65 tiles -> 9 blocks; writes h layout directly
  conv_wmma_kernel<16, 16, 4097, 1025, true>
      <<<dim3(9, BATCH, SFC), 256, 0, stream>>>(o2, w3, b3, hbuf);

  // fc1: grid (8 n-groups, 41 k-chunks), 8 waves/block, 4 acc per wave
  fc1_wmma_kernel<<<dim3(8, 41), 256, 0, stream>>>(hbuf, fw1, part);
  fc1_reduce_kernel<<<256, 256, 0, stream>>>(part, fb1, h1);

  // fc2: 4 waves, one block
  fc2_wmma_kernel<<<1, 128, 0, stream>>>(h1, fw2, fb2, (float*)d_out);
}
